// GNN_14027363189299
// MI455X (gfx1250) — compile-verified, matrix operand order checked by reading, not run on hardware
//
#include <hip/hip_runtime.h>

// ---------------- CDNA5 (gfx1250) bf16 WMMA GNN pipeline ----------------
// wave32, v_wmma_f32_16x16x32_bf16, ASYNC global->LDS weight staging
// (double-buffered, hidden behind WMMA), L2-resident f32 atomic segment-sum,
// global_prefetch on the ef stream.

#define SLOPE 0.01f

typedef __attribute__((ext_vector_type(8)))  __bf16 v8bf;
typedef __attribute__((ext_vector_type(16))) __bf16 v16bf;
typedef __attribute__((ext_vector_type(8)))  float  v8f;
typedef __attribute__((ext_vector_type(4)))  int    v4i;

static __device__ __forceinline__ float leaky_f(float v) {
  return v >= 0.0f ? v : SLOPE * v;
}

// ---- CDNA5 async global->LDS (ASYNCcnt-tracked) -------------------------
// Builtin parameter types (from clang diagnostic): (v4i AS1*, v4i AS3*, Ii, Ii)
static __device__ __forceinline__ void async_copy16(const void* g, void* s) {
#if __has_builtin(__builtin_amdgcn_global_load_async_to_lds_b128)
  __builtin_amdgcn_global_load_async_to_lds_b128(
      (__attribute__((address_space(1))) v4i*)(size_t)g,
      (__attribute__((address_space(3))) v4i*)(unsigned)(size_t)s,
      0, 0);
#else
  // LDS offset = low 32 bits of the flat shared address (ISA 10.2 aperture).
  unsigned lds_off = (unsigned)(size_t)s;
  asm volatile("global_load_async_to_lds_b128 %0, %1, off"
               :: "v"(lds_off), "v"(g) : "memory");
#endif
}

static __device__ __forceinline__ void wait_async0() {
#if __has_builtin(__builtin_amdgcn_s_wait_asynccnt)
  __builtin_amdgcn_s_wait_asynccnt(0);
#else
  asm volatile("s_wait_asynccnt 0x0" ::: "memory");
#endif
}

// Build a 16x32 (16-bit) WMMA A/B fragment for this lane.
// Lane L, half = L>>4.  VGPR0-3 hold K = 8*half .. +7, VGPR4-7 hold
// K = 16+8*half .. +7.  Caller passes p already offset by (k0 + 8*half).
static __device__ __forceinline__ v16bf frag_from_bf16(const __bf16* p) {
  v8bf lo = *(const v8bf*)(p);
  v8bf hi = *(const v8bf*)(p + 16);
  return __builtin_shufflevector(lo, hi, 0,1,2,3,4,5,6,7,8,9,10,11,12,13,14,15);
}

// Same fragment, f32 source (reduction buffer): convert in-register.
static __device__ __forceinline__ v16bf frag_from_f32(const float* p) {
  v8f lof = *(const v8f*)(p);
  v8f hif = *(const v8f*)(p + 16);
  v8bf lo = __builtin_convertvector(lof, v8bf);
  v8bf hi = __builtin_convertvector(hif, v8bf);
  return __builtin_shufflevector(lo, hi, 0,1,2,3,4,5,6,7,8,9,10,11,12,13,14,15);
}

// ---------------- helper kernels ----------------

__global__ __launch_bounds__(256)
void k_f32_to_bf16(const float* __restrict__ in, __bf16* __restrict__ out, long n) {
  long i = (long)blockIdx.x * 256 + threadIdx.x;
  if (i < n) out[i] = (__bf16)in[i];
}

// W [K][Ncol] row-major f32  ->  WT [Ncol][K] bf16 (transposed for B-fragments)
__global__ __launch_bounds__(256)
void k_transpose_to_bf16(const float* __restrict__ W, __bf16* __restrict__ WT,
                         int K, int Ncol) {
  long i = (long)blockIdx.x * 256 + threadIdx.x;
  long total = (long)K * Ncol;
  if (i < total) {
    int k = (int)(i / Ncol);
    int n = (int)(i % Ncol);
    WT[(long)n * K + k] = (__bf16)W[i];
  }
}

__global__ __launch_bounds__(256)
void k_active(const int* __restrict__ src, const int* __restrict__ dst,
              const int* __restrict__ ntype, float* __restrict__ act,
              unsigned* __restrict__ recv, int E) {
  int e = blockIdx.x * 256 + threadIdx.x;
  if (e < E) {
    int a = (ntype[src[e]] == 2) ? 1 : 0;   // TASK_TYPE == 2
    act[e] = (float)a;
    if (a) atomicOr(&recv[dst[e]], 1u);
  }
}

// ---------------- edge kernel ----------------
// Block = 256 threads = 8 waves, 128 edges/block.
// Wave w owns M-tile w (16 edges) x all 8 N-tiles (128 output cols).
// K = 320 = [h[src] 0:128 | h[dst] 128:256 | ef 256:320], stepped by 32.
__global__ __launch_bounds__(256)
void k_edge(const __bf16* __restrict__ h, const __bf16* __restrict__ efb,
            const __bf16* __restrict__ WeT, const float* __restrict__ act,
            const int* __restrict__ src, const int* __restrict__ dst,
            float* __restrict__ red, int E) {
  __shared__ __bf16 sB[2][128 * 40];   // double-buffered 32-K chunk of WeT

  const int tid  = threadIdx.x;
  const int lane = tid & 31;
  const int wave = tid >> 5;
  const int half = lane >> 4;
  const int l16  = lane & 15;

  const long ebase = (long)blockIdx.x * 128;
  int e_row = (int)(ebase + wave * 16 + l16);
  if (e_row >= E) e_row = E - 1;                 // clamped read; stores guarded
  const int s_row = src[e_row];
  const int d_row = dst[e_row];

  const __bf16* hs = h   + (size_t)s_row * 128;
  const __bf16* hd = h   + (size_t)d_row * 128;
  const __bf16* ee = efb + (size_t)e_row * 64;
  __builtin_prefetch(ee, 0, 0);                  // global_prefetch_b8 (ef stream)

  v8f acc[8];
#pragma unroll
  for (int t = 0; t < 8; ++t)
#pragma unroll
    for (int j = 0; j < 8; ++j) acc[t][j] = 0.0f;

  const int c_col = tid >> 1;                    // staging: col, 32B half-chunk
  const int c_h   = tid & 1;

  // async-stage one 32-K chunk of WeT into sB[buf]
  auto stage = [&](int ks, int buf) {
    const __bf16* g = WeT + (size_t)c_col * 320 + ks * 32 + c_h * 16;
    __bf16* s = &sB[buf][c_col * 40 + c_h * 16];
    async_copy16(g, s);
    async_copy16(g + 8, s + 8);
  };

  stage(0, 0);
  wait_async0();
  __syncthreads();

  for (int ks = 0; ks < 10; ++ks) {
    const int buf = ks & 1;
    if (ks + 1 < 10) stage(ks + 1, buf ^ 1);     // overlap copy with WMMA

    const int k0 = ks * 32;
    const __bf16* ap;
    if (k0 < 128)      ap = hs + k0;
    else if (k0 < 256) ap = hd + (k0 - 128);
    else               ap = ee + (k0 - 256);
    v16bf a = frag_from_bf16(ap + half * 8);

#pragma unroll
    for (int t = 0; t < 8; ++t) {
      v16bf b = frag_from_bf16(&sB[buf][(t * 16 + l16) * 40 + half * 8]);
      acc[t] = __builtin_amdgcn_wmma_f32_16x16x32_bf16(
          false, a, false, b, (short)0, acc[t], false, false);
    }

    wait_async0();                               // own async LDS writes landed
    __syncthreads();                             // everyone's chunk visible
  }

  // Epilogue: C layout row M = j + 8*half, col N = t*16 + l16.
  // leaky -> *active -> atomic scatter-add into red[dst] (L2-resident).
#pragma unroll
  for (int j = 0; j < 8; ++j) {
    long e = ebase + wave * 16 + j + half * 8;
    if (e >= E) continue;
    float a_e = act[e];
    if (a_e == 0.0f) continue;                   // ~75% of edges skip atomics
    int dnode = dst[e];
    float* rrow = red + (size_t)dnode * 128 + l16;
#pragma unroll
    for (int t = 0; t < 8; ++t) {
      float v = leaky_f(acc[t][j]) * a_e;
      atomicAdd(rrow + t * 16, v);
    }
  }
}

// ---------------- node kernel ----------------
// Block = 256 threads = 8 waves, 128 nodes/block.  K = 256 = [h | red(f32)].
__global__ __launch_bounds__(256)
void k_node(const __bf16* __restrict__ h, const float* __restrict__ red,
            const __bf16* __restrict__ WnT, const unsigned* __restrict__ recv,
            __bf16* __restrict__ hout, float* __restrict__ fout, int N) {
  __shared__ __bf16 sB[2][128 * 40];

  const int tid  = threadIdx.x;
  const int lane = tid & 31;
  const int wave = tid >> 5;
  const int half = lane >> 4;
  const int l16  = lane & 15;

  const long nbase = (long)blockIdx.x * 128;
  int n_row = (int)(nbase + wave * 16 + l16);
  if (n_row >= N) n_row = N - 1;
  const __bf16* hr = h   + (size_t)n_row * 128;
  const float*  rr = red + (size_t)n_row * 128;

  v8f acc[8];
#pragma unroll
  for (int t = 0; t < 8; ++t)
#pragma unroll
    for (int j = 0; j < 8; ++j) acc[t][j] = 0.0f;

  const int c_col = tid >> 1;
  const int c_h   = tid & 1;

  auto stage = [&](int ks, int buf) {
    const __bf16* g = WnT + (size_t)c_col * 256 + ks * 32 + c_h * 16;
    __bf16* s = &sB[buf][c_col * 40 + c_h * 16];
    async_copy16(g, s);
    async_copy16(g + 8, s + 8);
  };

  stage(0, 0);
  wait_async0();
  __syncthreads();

  for (int ks = 0; ks < 8; ++ks) {
    const int buf = ks & 1;
    if (ks + 1 < 8) stage(ks + 1, buf ^ 1);

    const int k0 = ks * 32;
    v16bf a;
    if (k0 < 128) a = frag_from_bf16(hr + k0 + half * 8);
    else          a = frag_from_f32(rr + (k0 - 128) + half * 8);

#pragma unroll
    for (int t = 0; t < 8; ++t) {
      v16bf b = frag_from_bf16(&sB[buf][(t * 16 + l16) * 40 + half * 8]);
      acc[t] = __builtin_amdgcn_wmma_f32_16x16x32_bf16(
          false, a, false, b, (short)0, acc[t], false, false);
    }

    wait_async0();
    __syncthreads();
  }

#pragma unroll
  for (int j = 0; j < 8; ++j) {
    long n = nbase + wave * 16 + j + half * 8;
    if (n >= N) continue;
    float rv = recv[n] ? 1.0f : 0.0f;
#pragma unroll
    for (int t = 0; t < 8; ++t) {
      float v = leaky_f(acc[t][j]) * rv;
      long o = n * 128 + t * 16 + l16;
      hout[o] = (__bf16)v;
      if (fout) fout[o] = v;
    }
  }
}

// ---------------- host launcher ----------------

extern "C" void kernel_launch(void* const* d_in, const int* in_sizes, int n_in,
                              void* d_out, int out_size, void* d_ws, size_t ws_size,
                              hipStream_t stream) {
  const float* nf    = (const float*)d_in[0];
  const float* ef    = (const float*)d_in[1];
  const int*   src   = (const int*)d_in[2];
  const int*   dst   = (const int*)d_in[3];
  const int*   ntype = (const int*)d_in[4];
  const float* We[3] = { (const float*)d_in[5], (const float*)d_in[7], (const float*)d_in[9]  };
  const float* Wn[3] = { (const float*)d_in[6], (const float*)d_in[8], (const float*)d_in[10] };

  const int N = in_sizes[0] / 128;   // 50000
  const int E = in_sizes[2];         // 800000
  const long NH = (long)N * 128;
  const long EH = (long)E * 64;

  // carve workspace (256B aligned)
  char* p = (char*)d_ws;
  auto carve = [&](size_t bytes) {
    char* r = p; p += (bytes + 255) & ~(size_t)255; return r;
  };
  __bf16*   hA     = (__bf16*)carve(NH * 2);
  __bf16*   hB     = (__bf16*)carve(NH * 2);
  __bf16*   efb    = (__bf16*)carve(EH * 2);
  __bf16*   WeT[3], *WnT[3];
  for (int i = 0; i < 3; ++i) WeT[i] = (__bf16*)carve((size_t)128 * 320 * 2);
  for (int i = 0; i < 3; ++i) WnT[i] = (__bf16*)carve((size_t)128 * 256 * 2);
  float*    red    = (float*)carve(NH * 4);
  float*    act    = (float*)carve((size_t)E * 4);
  unsigned* recv   = (unsigned*)carve((size_t)N * 4);
  (void)ws_size; (void)n_in; (void)out_size;

  // one-time conversions
  k_f32_to_bf16<<<(unsigned)((NH + 255) / 256), 256, 0, stream>>>(nf, hA, NH);
  k_f32_to_bf16<<<(unsigned)((EH + 255) / 256), 256, 0, stream>>>(ef, efb, EH);
  for (int i = 0; i < 3; ++i) {
    k_transpose_to_bf16<<<(320 * 128 + 255) / 256, 256, 0, stream>>>(We[i], WeT[i], 320, 128);
    k_transpose_to_bf16<<<(256 * 128 + 255) / 256, 256, 0, stream>>>(Wn[i], WnT[i], 256, 128);
  }
  (void)hipMemsetAsync(recv, 0, (size_t)N * 4, stream);
  k_active<<<(E + 255) / 256, 256, 0, stream>>>(src, dst, ntype, act, recv, E);

  // 3 GNN layers
  __bf16* hc = hA;
  __bf16* hn = hB;
  const unsigned eg = (unsigned)((E + 127) / 128);
  const unsigned ng = (unsigned)((N + 127) / 128);
  for (int i = 0; i < 3; ++i) {
    (void)hipMemsetAsync(red, 0, NH * 4, stream);
    k_edge<<<eg, 256, 0, stream>>>(hc, efb, WeT[i], act, src, dst, red, E);
    k_node<<<ng, 256, 0, stream>>>(hc, red, WnT[i], recv, hn,
                                   (i == 2) ? (float*)d_out : nullptr, N);
    __bf16* t = hc; hc = hn; hn = t;
  }
}
